// LSTMClassifier_14388140441825
// MI455X (gfx1250) — compile-verified, matrix operand order checked by reading, not run on hardware
//
#include <hip/hip_runtime.h>

// Problem sizes (match reference)
#define B_ 512
#define T_ 512
#define D_ 64
#define H_ 128
#define G_ 512   // 4*H gates
#define OUT_ 16
#define BT_ 16       // batch rows per workgroup (one M-tile)
#define THREADS_ 512 // 16 waves of 32
#define NT_ 2        // 16-wide N tiles per wave (16 waves * 32 cols = 512 gate cols)

typedef __attribute__((ext_vector_type(16))) _Float16 v16h;
typedef __attribute__((ext_vector_type(8)))  _Float16 v8h;
typedef __attribute__((ext_vector_type(8)))  float    v8f;

// ---- fast, branch-free activations ----
__device__ __forceinline__ float fast_tanh(float x) {
#if __has_builtin(__builtin_amdgcn_tanhf)
  return __builtin_amdgcn_tanhf(x);           // v_tanh_f32 (gfx1250 TRANS op)
#else
  float e = __builtin_amdgcn_exp2f(x * 2.8853900817779268f);
  return 1.0f - 2.0f * __builtin_amdgcn_rcpf(e + 1.0f);
#endif
}

// Load a 16x32 f16 WMMA A/B fragment from row-major [16 rows x stride] memory.
// ISA layout (CDNA5 ISA): lanes 0-15 hold K 0..7 and 16..23, lanes 16-31 hold
// K 8..15 and 24..31 -> two contiguous 16B loads per lane.
__device__ __forceinline__ v16h ldfrag_g(const _Float16* __restrict__ base,
                                         int stride, int koff, int lane) {
  int r  = lane & 15;
  int hi = lane >> 4;
  const _Float16* p = base + (size_t)r * stride + koff + hi * 8;
  v8h a = *(const v8h*)p;
  v8h b = *(const v8h*)(p + 16);
  return __builtin_shufflevector(a, b, 0,1,2,3,4,5,6,7,8,9,10,11,12,13,14,15);
}

__global__ void cvt_f32_to_f16(const float* __restrict__ s, _Float16* __restrict__ d, int n) {
  int i = blockIdx.x * blockDim.x + threadIdx.x;
  if (i < n) d[i] = (_Float16)s[i];
}

__global__ void add_vec(const float* __restrict__ a, const float* __restrict__ b,
                        float* __restrict__ o, int n) {
  int i = blockIdx.x * blockDim.x + threadIdx.x;
  if (i < n) o[i] = a[i] + b[i];
}

// One LSTM layer, fused input projection + recurrence.
// xseq: (B, T, KIN) f16; Wih: (G, KIN) f16; Whh: (G, H) f16; bias = bih+bhh (G) f32.
// hseq_out: (B, T, H) f16 (layer 0) or null; hlast_out: (B, H) f32 (layer 1) or null.
template<int KIN>
__global__ __launch_bounds__(THREADS_, 1)
void lstm_layer_kernel(const _Float16* __restrict__ xseq,
                       const _Float16* __restrict__ Wih,
                       const _Float16* __restrict__ Whh,
                       const float*    __restrict__ bias,
                       _Float16*       __restrict__ hseq_out,
                       float*          __restrict__ hlast_out)
{
  constexpr int KX = KIN / 32;

  __shared__ _Float16 h_lds[BT_][H_ + 8];   // h_{t-1}, padded (272B row, bank-spread)
  __shared__ float    g_lds[BT_][G_ + 8];   // activated gates exchange buffer

  const int tid   = threadIdx.x;
  const int lane  = tid & 31;
  const int wave  = tid >> 5;               // 0..15
  const int hi    = lane >> 4;              // half-wave select
  const int b0    = blockIdx.x * BT_;
  const int ncol0 = wave * (NT_ * 16);      // this wave's first gate column

  // h_0 = 0
  for (int i = tid; i < BT_ * (H_ + 8); i += THREADS_)
    ((_Float16*)h_lds)[i] = (_Float16)0.0f;

  // Preload weight B-fragments: resident in VGPRs for the whole T loop.
  v16h Bx[NT_][KX];
  v16h Bh[NT_][4];
  float bv[NT_];
  #pragma unroll
  for (int n = 0; n < NT_; ++n) {
    const int nc = ncol0 + n * 16;
    bv[n] = bias[nc + (lane & 15)];
    #pragma unroll
    for (int kt = 0; kt < KX; ++kt)
      Bx[n][kt] = ldfrag_g(Wih + (size_t)nc * KIN, KIN, kt * 32, lane);
    #pragma unroll
    for (int kt = 0; kt < 4; ++kt)
      Bh[n][kt] = ldfrag_g(Whh + (size_t)nc * H_, H_, kt * 32, lane);
  }

  // Wave-uniform activation: gate g (cols 256..383) is tanh, others sigmoid.
  // sigmoid(x) = 0.5 + 0.5*tanh(0.5*x), so both reduce to one v_tanh_f32 with
  // uniform pre/post/offset constants -> zero branches in the hot loop.
  const bool  is_tanh = ((wave >> 2) == 2);
  const float act_pre  = is_tanh ? 1.0f : 0.5f;
  const float act_post = is_tanh ? 1.0f : 0.5f;
  const float act_off  = is_tanh ? 0.0f : 0.5f;

  // Per-thread cell state: q = tid*4+e -> (r = q>>7, j = q&127)
  float c[4] = {0.f, 0.f, 0.f, 0.f};

  const _Float16* xbase = xseq + (size_t)b0 * T_ * KIN;

  __syncthreads();

  for (int t = 0; t < T_; ++t) {
    // ---- A fragments: x_t (global, L2-resident) and h_{t-1} (LDS) ----
    v16h xa[KX];
    #pragma unroll
    for (int kt = 0; kt < KX; ++kt)
      xa[kt] = ldfrag_g(xbase + (size_t)t * KIN, T_ * KIN, kt * 32, lane);
    if (t + 1 < T_)
      __builtin_prefetch(xbase + (size_t)(t + 1) * KIN + (size_t)(lane & 15) * (T_ * KIN), 0, 1);

    v16h ha[4];
    #pragma unroll
    for (int kt = 0; kt < 4; ++kt) {
      const _Float16* p = &h_lds[lane & 15][kt * 32 + hi * 8];
      v8h a = *(const v8h*)p;
      v8h b = *(const v8h*)(p + 16);
      ha[kt] = __builtin_shufflevector(a, b, 0,1,2,3,4,5,6,7,8,9,10,11,12,13,14,15);
    }

    // ---- gates = bias + x_t @ Wih^T + h_{t-1} @ Whh^T (f32 accum WMMA) ----
    #pragma unroll
    for (int n = 0; n < NT_; ++n) {
      v8f acc;
      #pragma unroll
      for (int m = 0; m < 8; ++m) acc[m] = bv[n];  // bias is per-column (N = lane&15)
      #pragma unroll
      for (int kt = 0; kt < KX; ++kt)
        acc = __builtin_amdgcn_wmma_f32_16x16x32_f16(false, xa[kt], false, Bx[n][kt],
                                                     (short)0, acc, false, false);
      #pragma unroll
      for (int kt = 0; kt < 4; ++kt)
        acc = __builtin_amdgcn_wmma_f32_16x16x32_f16(false, ha[kt], false, Bh[n][kt],
                                                     (short)0, acc, false, false);

      // Activate (mul -> v_tanh_f32 -> fma, branch-free) and scatter C-fragment
      // to LDS: element m holds row M = m + 8*hi, column N = ncol0+n*16+(lane&15).
      const int col = ncol0 + n * 16 + (lane & 15);
      #pragma unroll
      for (int m = 0; m < 8; ++m) {
        float a = act_off + act_post * fast_tanh(acc[m] * act_pre);
        g_lds[m + 8 * hi][col] = a;
      }
    }
    __syncthreads();

    // ---- c/h update: 2048 (r,j) cells, 4 per thread, c in registers ----
    #pragma unroll
    for (int e = 0; e < 4; ++e) {
      int q = tid * 4 + e;
      int r = q >> 7;
      int j = q & (H_ - 1);
      float iv = g_lds[r][j];
      float fv = g_lds[r][H_ + j];
      float gv = g_lds[r][2 * H_ + j];
      float ov = g_lds[r][3 * H_ + j];
      c[e] = fv * c[e] + iv * gv;
      float hv = ov * fast_tanh(c[e]);
      h_lds[r][j] = (_Float16)hv;
      if (hseq_out)
        hseq_out[(size_t)(b0 + r) * T_ * H_ + (size_t)t * H_ + j] = (_Float16)hv;
      if (hlast_out && t == T_ - 1)
        hlast_out[(b0 + r) * H_ + j] = hv;
    }
    __syncthreads();
  }
}

// out = (h_last @ Wfc0^T + bfc0) @ Wfc1^T + bfc1 ; tiny, one wave per batch row.
__global__ void fc_kernel(const float* __restrict__ h1, const float* __restrict__ Wfc0,
                          const float* __restrict__ bfc0, const float* __restrict__ Wfc1,
                          const float* __restrict__ bfc1, float* __restrict__ out) {
  int b = blockIdx.x;
  int tid = threadIdx.x;  // 32 threads
  __shared__ float tmp[32];
  float s = bfc0[tid];
  for (int k = 0; k < H_; ++k) s += h1[b * H_ + k] * Wfc0[tid * H_ + k];
  tmp[tid] = s;
  __syncthreads();
  if (tid < OUT_) {
    float o = bfc1[tid];
    for (int k = 0; k < 32; ++k) o += tmp[k] * Wfc1[tid * 32 + k];
    out[b * OUT_ + tid] = o;
  }
}

extern "C" void kernel_launch(void* const* d_in, const int* in_sizes, int n_in,
                              void* d_out, int out_size, void* d_ws, size_t ws_size,
                              hipStream_t stream) {
  (void)in_sizes; (void)n_in; (void)out_size; (void)ws_size;
  const float* x    = (const float*)d_in[0];
  const float* Wih0 = (const float*)d_in[1];
  const float* Whh0 = (const float*)d_in[2];
  const float* bih0 = (const float*)d_in[3];
  const float* bhh0 = (const float*)d_in[4];
  const float* Wih1 = (const float*)d_in[5];
  const float* Whh1 = (const float*)d_in[6];
  const float* bih1 = (const float*)d_in[7];
  const float* bhh1 = (const float*)d_in[8];
  const float* Wfc0 = (const float*)d_in[9];
  const float* bfc0 = (const float*)d_in[10];
  const float* Wfc1 = (const float*)d_in[11];
  const float* bfc1 = (const float*)d_in[12];

  char* p = (char*)d_ws;
  auto alloc = [&](size_t bytes) {
    char* r = p;
    p += (bytes + 255) & ~(size_t)255;
    return r;
  };
  _Float16* x_h    = (_Float16*)alloc((size_t)B_ * T_ * D_ * 2);  // 33.5 MB
  _Float16* h0seq  = (_Float16*)alloc((size_t)B_ * T_ * H_ * 2);  // 67 MB
  _Float16* wih0_h = (_Float16*)alloc((size_t)G_ * D_ * 2);
  _Float16* whh0_h = (_Float16*)alloc((size_t)G_ * H_ * 2);
  _Float16* wih1_h = (_Float16*)alloc((size_t)G_ * H_ * 2);
  _Float16* whh1_h = (_Float16*)alloc((size_t)G_ * H_ * 2);
  float*    b0     = (float*)alloc((size_t)G_ * 4);
  float*    b1     = (float*)alloc((size_t)G_ * 4);
  float*    h1last = (float*)alloc((size_t)B_ * H_ * 4);

  auto cvt = [&](const float* s, _Float16* d, int n) {
    cvt_f32_to_f16<<<(n + 255) / 256, 256, 0, stream>>>(s, d, n);
  };
  cvt(x, x_h, B_ * T_ * D_);
  cvt(Wih0, wih0_h, G_ * D_);
  cvt(Whh0, whh0_h, G_ * H_);
  cvt(Wih1, wih1_h, G_ * H_);
  cvt(Whh1, whh1_h, G_ * H_);
  add_vec<<<2, 256, 0, stream>>>(bih0, bhh0, b0, G_);
  add_vec<<<2, 256, 0, stream>>>(bih1, bhh1, b1, G_);

  lstm_layer_kernel<D_><<<B_ / BT_, THREADS_, 0, stream>>>(x_h, wih0_h, whh0_h, b0,
                                                           h0seq, nullptr);
  lstm_layer_kernel<H_><<<B_ / BT_, THREADS_, 0, stream>>>(h0seq, wih1_h, whh1_h, b1,
                                                           nullptr, h1last);
  fc_kernel<<<B_, 32, 0, stream>>>(h1last, Wfc0, bfc0, Wfc1, bfc1, (float*)d_out);
}
